// DPOTNet_81793357185869
// MI455X (gfx1250) — matrix operand order repaired and checked by reading, not compiled, not run on hardware
//
#include <hip/hip_runtime.h>
#include <hip/hip_bf16.h>

// ---------------------------------------------------------------------------
// DPOT forward for MI455X (gfx1250).  All dense contractions run on
// v_wmma_f32_16x16x32_f16 (f16 in / f32 accumulate).  Weights are pre-packed
// into WMMA B-fragment order (f16, contiguous 32B per lane).  Each GEMM wave
// computes a 16x64 strip: per K-step, 1 shared A fragment (4x b128 + cvt) and
// 4 packed-B tiles feed 4 independent WMMAs.  Truncated rFFT2 in the AFNO
// blocks is expressed as DFT GEMMs.
// ---------------------------------------------------------------------------

typedef __attribute__((ext_vector_type(16))) _Float16 v16h;
typedef __attribute__((ext_vector_type(8)))  float    v8f;

#define WMMA_F16(a, b, c) \
  __builtin_amdgcn_wmma_f32_16x16x32_f16(false, (a), false, (b), (short)0, (c), false, false)

__device__ __forceinline__ float gelu_f(float x) {
  return 0.5f * x * (1.0f + erff(x * 0.70710678118654752440f));
}

// ---------------------------------------------------------------------------
// Fragment helpers (layouts per ISA 7.12.2).
// A (16x32 MxK): lane m = lane&15; lanes<16 hold K 0-7,16-23; lanes>=16 hold
// K 8-15,24-31.  With base = (lane<16?0:8), lane reads K runs [base,base+8)
// and [base+16,base+24) -> two contiguous float4 pairs.
// ---------------------------------------------------------------------------
__device__ __forceinline__ v16h cvt_fragA(const float* p) {
  const float4 a0 = *(const float4*)(p);
  const float4 a1 = *(const float4*)(p + 4);
  const float4 a2 = *(const float4*)(p + 16);
  const float4 a3 = *(const float4*)(p + 20);
  v16h f;
  f[0] = (_Float16)a0.x;  f[1] = (_Float16)a0.y;  f[2] = (_Float16)a0.z;  f[3] = (_Float16)a0.w;
  f[4] = (_Float16)a1.x;  f[5] = (_Float16)a1.y;  f[6] = (_Float16)a1.z;  f[7] = (_Float16)a1.w;
  f[8] = (_Float16)a2.x;  f[9] = (_Float16)a2.y;  f[10] = (_Float16)a2.z; f[11] = (_Float16)a2.w;
  f[12] = (_Float16)a3.x; f[13] = (_Float16)a3.y; f[14] = (_Float16)a3.z; f[15] = (_Float16)a3.w;
  return f;
}

__device__ __forceinline__ v16h cvt_fragA_s(const float* p, float sgn) {
  const float4 a0 = *(const float4*)(p);
  const float4 a1 = *(const float4*)(p + 4);
  const float4 a2 = *(const float4*)(p + 16);
  const float4 a3 = *(const float4*)(p + 20);
  v16h f;
  f[0] = (_Float16)(a0.x * sgn);  f[1] = (_Float16)(a0.y * sgn);
  f[2] = (_Float16)(a0.z * sgn);  f[3] = (_Float16)(a0.w * sgn);
  f[4] = (_Float16)(a1.x * sgn);  f[5] = (_Float16)(a1.y * sgn);
  f[6] = (_Float16)(a1.z * sgn);  f[7] = (_Float16)(a1.w * sgn);
  f[8] = (_Float16)(a2.x * sgn);  f[9] = (_Float16)(a2.y * sgn);
  f[10] = (_Float16)(a2.z * sgn); f[11] = (_Float16)(a2.w * sgn);
  f[12] = (_Float16)(a3.x * sgn); f[13] = (_Float16)(a3.y * sgn);
  f[14] = (_Float16)(a3.z * sgn); f[15] = (_Float16)(a3.w * sgn);
  return f;
}

// Constant DFT matrix fragments (fully in-range, contiguous).
__device__ __forceinline__ v16h frag_const16x32(const float* Mt, int lane) {
  return cvt_fragA(Mt + (size_t)(lane & 15) * 32 + ((lane < 16) ? 0 : 8));
}
__device__ __forceinline__ v16h frag_const32x32(const float* Mt, int m0, int lane) {
  return cvt_fragA(Mt + (size_t)(m0 + (lane & 15)) * 32 + ((lane < 16) ? 0 : 8));
}

// A fragment from f16 LDS tile (16 x lda).
__device__ __forceinline__ v16h load_fragA_h(const _Float16* A, int lda,
                                             int k0, int lane, float sgn) {
  v16h f;
  int m = lane & 15;
  int base = (lane < 16) ? 0 : 8;
#pragma unroll
  for (int j = 0; j < 16; ++j) {
    int k = k0 + base + ((j < 8) ? j : (j + 8));
    f[j] = (_Float16)((float)A[m * lda + k] * sgn);
  }
  return f;
}

// Strided f32 B fragment (KxN slice), fully in range, branchless.
// lane n = lane&15; lanes<16 hold K rows 0-7,16-23, lanes>=16 hold 8-15,24-31.
__device__ __forceinline__ v16h load_fragB_s(const float* Bm, size_t ldb,
                                             int n0, int lane) {
  const float* p = Bm + (size_t)((lane < 16) ? 0 : 8) * ldb + n0 + (lane & 15);
  v16h f;
#pragma unroll
  for (int j = 0; j < 16; ++j) f[j] = (_Float16)p[(size_t)((j < 8) ? j : (j + 8)) * ldb];
  return f;
}

// Same, but rows k >= Klim selected to zero (loads stay unconditional;
// over-read stays inside the workspace arena).
__device__ __forceinline__ v16h load_fragB_sk(const float* Bm, size_t ldb,
                                              int n0, int lane, int Klim) {
  int kb = (lane < 16) ? 0 : 8;
  const float* p = Bm + (size_t)kb * ldb + n0 + (lane & 15);
  v16h f;
#pragma unroll
  for (int j = 0; j < 16; ++j) {
    int jj = (j < 8) ? j : (j + 8);
    float v = p[(size_t)jj * ldb];
    f[j] = (_Float16)((kb + jj < Klim) ? v : 0.0f);
  }
  return f;
}

// C/D store: lane n = lane&15; lanes<16 rows m0..m0+7, lanes>=16 rows m0+8..15.
__device__ __forceinline__ void store_tile(float* C, size_t ldc, int m0, int n0,
                                           int lane, v8f acc) {
  int n = n0 + (lane & 15);
  int mb = m0 + ((lane < 16) ? 0 : 8);
#pragma unroll
  for (int r = 0; r < 8; ++r) C[(size_t)(mb + r) * ldc + n] = acc[r];
}

// ---------------------------------------------------------------------------
// Weight packing: f32 (K x N) row-major -> f16 WMMA B fragments.
// N-tile grid padded to a multiple of 4 (zero-filled) so the 16x64 GEMM
// strip never reads OOB.  Tile (kt,nt): 512 halves at ((kt*ntn4)+nt)*512.
// ---------------------------------------------------------------------------
__host__ __device__ static inline int ntn4_of(int N) {
  return (((N + 15) >> 4) + 3) & ~3;
}

__global__ void k_packB(const float* __restrict__ Bm, _Float16* __restrict__ out,
                        int K, int N, int ldb) {
  int ntn4 = ntn4_of(N), ktn = (K + 31) >> 5;
  int total = ktn * ntn4 * 512;
  int idx = blockIdx.x * 256 + threadIdx.x;
  if (idx >= total) return;
  int j = idx & 15, lane = (idx >> 4) & 31, tile = idx >> 9;
  int kt = tile / ntn4, nt = tile - kt * ntn4;
  int n = (nt << 4) + (lane & 15);
  int k = (kt << 5) + ((lane < 16) ? 0 : 8) + ((j < 8) ? j : (j + 8));
  float v = (k < K && n < N) ? Bm[(size_t)k * ldb + n] : 0.0f;
  out[idx] = (_Float16)v;
}

// dec_w (8,8,32,512) -> packed B of DecW^T (512 x 2048): B[k][n] = dw[n*512+k]
__global__ void k_pack_decw(const float* __restrict__ dw, _Float16* __restrict__ out) {
  int idx = blockIdx.x * 256 + threadIdx.x;      // 16*128*512
  if (idx >= 1048576) return;
  int j = idx & 15, lane = (idx >> 4) & 31, tile = idx >> 9;
  int nt = tile & 127;
  int kt = tile >> 7;
  int n = (nt << 4) + (lane & 15);
  int k = (kt << 5) + ((lane < 16) ? 0 : 8) + ((j < 8) ? j : (j + 8));
  out[idx] = (_Float16)dw[(size_t)n * 512 + k];
}

// AFNO weight blob (12,2,8,64,64) -> 192 packed 64x64 matrices (8 tiles each).
__global__ void k_pack_afno(const float* __restrict__ w, _Float16* __restrict__ out) {
  int idx = blockIdx.x * 256 + threadIdx.x;      // 192*4096
  if (idx >= 192 * 4096) return;
  int j = idx & 15, lane = (idx >> 4) & 31, tile = (idx >> 9) & 7, mi = idx >> 12;
  int kt = tile >> 2, nt = tile & 3;
  int n = (nt << 4) + (lane & 15);
  int k = (kt << 5) + ((lane < 16) ? 0 : 8) + ((j < 8) ? j : (j + 8));
  out[idx] = (_Float16)w[((size_t)mi * 64 + k) * 64 + n];
}

__global__ void k_zero(float* __restrict__ p, int n) {
  int i = blockIdx.x * 256 + threadIdx.x;
  if (i < n) p[i] = 0.0f;
}

// ---------------------------------------------------------------------------
// Packed-B GEMM, 16x64 strip per wave (4 accumulators, shared A fragment):
// C = [gelu](A@Bpk + bias) [+ extra[(m % ermod)*eld + n]]
// A needs lda % 4 == 0 and Ktiles*32 readable columns (zero-padded if ragged).
// ---------------------------------------------------------------------------
__global__ void k_gemm_pk(const float* __restrict__ A, const _Float16* __restrict__ Bpk,
                          const float* __restrict__ bias, const float* __restrict__ extra,
                          int ermod, int eld, float* __restrict__ C,
                          int M, int N, int Ktiles, int lda, int ldc, int fgelu) {
  int wave = blockIdx.x * 8 + (threadIdx.x >> 5);
  int lane = threadIdx.x & 31;
  int ntm = (M + 15) >> 4;
  int ntn4 = ntn4_of(N);
  int ngrp = ntn4 >> 2;
  if (wave >= ntm * ngrp) return;
  int m0 = (wave / ngrp) << 4;
  int ng = wave % ngrp;
  int mrow = m0 + (lane & 15);
  if (mrow >= M) mrow = 0;                        // clamp, masked at store
  const float* ap = A + (size_t)mrow * lda + ((lane < 16) ? 0 : 8);
  const _Float16* bp = Bpk + ((size_t)(ng * 4) * 32 + lane) * 16;
  const size_t bstride = (size_t)ntn4 * 512;
  v8f acc0 = {}, acc1 = {}, acc2 = {}, acc3 = {};
  for (int kt = 0; kt < Ktiles; ++kt) {
    __builtin_prefetch(ap + 32, 0, 1);            // speculative: safe on gfx1250
    v16h fa = cvt_fragA(ap);
    v16h fb0 = *(const v16h*)(bp);
    v16h fb1 = *(const v16h*)(bp + 512);
    v16h fb2 = *(const v16h*)(bp + 1024);
    v16h fb3 = *(const v16h*)(bp + 1536);
    acc0 = WMMA_F16(fa, fb0, acc0);
    acc1 = WMMA_F16(fa, fb1, acc1);
    acc2 = WMMA_F16(fa, fb2, acc2);
    acc3 = WMMA_F16(fa, fb3, acc3);
    ap += 32;
    bp += bstride;
  }
  int mb = m0 + ((lane < 16) ? 0 : 8);
  v8f accs[4] = {acc0, acc1, acc2, acc3};
#pragma unroll
  for (int s2 = 0; s2 < 4; ++s2) {
    int n = ((ng * 4 + s2) << 4) + (lane & 15);
    if (n >= N) continue;
    float bv = bias ? bias[n] : 0.0f;
#pragma unroll
    for (int r = 0; r < 8; ++r) {
      int m = mb + r;
      if (m >= M) continue;
      float v = accs[s2][r] + bv;
      if (fgelu) v = gelu_f(v);
      if (extra) v += extra[(size_t)(m % ermod) * eld + n];
      C[(size_t)m * ldc + n] = v;
    }
  }
}

// ---------------------------------------------------------------------------
// GroupNorm over (h,w,channels-in-group) per (batch, group): 1024*64 values.
// ---------------------------------------------------------------------------
__global__ void k_groupnorm(const float* __restrict__ X, const float* __restrict__ sc,
                            const float* __restrict__ bi, float* __restrict__ Y) {
  int b = blockIdx.x >> 3, g = blockIdx.x & 7;
  __shared__ float s_sum[256], s_sq[256];
  const float* base = X + (size_t)b * 1024 * 512 + g * 64;
  float sum = 0.f, sq = 0.f;
  for (int i = threadIdx.x; i < 65536; i += 256) {
    float v = base[(size_t)(i >> 6) * 512 + (i & 63)];
    sum += v; sq += v * v;
  }
  s_sum[threadIdx.x] = sum; s_sq[threadIdx.x] = sq;
  __syncthreads();
  for (int s = 128; s > 0; s >>= 1) {
    if (threadIdx.x < s) { s_sum[threadIdx.x] += s_sum[threadIdx.x + s];
                           s_sq[threadIdx.x]  += s_sq[threadIdx.x + s]; }
    __syncthreads();
  }
  float mu = s_sum[0] * (1.0f / 65536.0f);
  float var = s_sq[0] * (1.0f / 65536.0f) - mu * mu;
  float rstd = rsqrtf(var + 1e-5f);
  float* out = Y + (size_t)b * 1024 * 512 + g * 64;
  for (int i = threadIdx.x; i < 65536; i += 256) {
    int r = i >> 6, c = i & 63, ch = g * 64 + c;
    float v = base[(size_t)r * 512 + c];
    out[(size_t)r * 512 + c] = (v - mu) * rstd * sc[ch] + bi[ch];
  }
}

// ---------------------------------------------------------------------------
// DFT constants (workspace floats):
//   0: C1c(512) 512: C1s=-sin(512) 1024: C2c(512) 1536: C2s=+sin(512)
//   2048: C2sn=-sin(512) 2560: Erp(1024) 3584: Eip(1024) 4608: Einp(1024)
//   5632: Arp(1024) 6656: Aip(1024)   (32x32 zero-padded beyond 16 modes)
// ---------------------------------------------------------------------------
__global__ void k_init_dft(float* __restrict__ M) {
  int i = threadIdx.x;
  const float inv = 0.17677669529663688f;        // 1/sqrt(32)
  const float w32 = 0.19634954084936207f;        // 2*pi/32
  if (i < 512) {                                 // 16x32 [m][h]
    int m = i >> 5, h = i & 31;
    float a = w32 * (float)(m * h);
    M[i]        =  cosf(a) * inv;
    M[512 + i]  = -sinf(a) * inv;
    M[1024 + i] =  cosf(a) * inv;
    M[1536 + i] =  sinf(a) * inv;
    M[2048 + i] = -sinf(a) * inv;
  }
  if (i < 1024) {                                // 32x32 padded [h][m], m<16 valid
    int h = i >> 5, m = i & 31;
    bool valid = m < 16;
    float a = w32 * (float)(m * h);
    float er = valid ? cosf(a) * inv : 0.0f;
    float ei = valid ? sinf(a) * inv : 0.0f;
    M[2560 + i] = er;
    M[3584 + i] = ei;
    M[4608 + i] = -ei;
    M[5632 + i] = !valid ? 0.0f : ((m == 0) ? inv : 2.0f * cosf(a) * inv);
    M[6656 + i] = !valid ? 0.0f : ((m == 0) ? 0.0f : -2.0f * sinf(a) * inv);
  }
}

// Forward DFT along h: Y[b,m,w,c] = sum_h C1[m,h] x[b,h,w,c].
__global__ void k_dft_rows(const float* __restrict__ X, const float* __restrict__ DM,
                           float* __restrict__ Yr, float* __restrict__ Yi) {
  int wave = blockIdx.x * 8 + (threadIdx.x >> 5);
  int lane = threadIdx.x & 31;
  int b = wave >> 10, tn = wave & 1023;
  if (b >= 8) return;
  const float* Xb = X + (size_t)b * 32 * 16384;
  v16h ac = frag_const16x32(DM, lane);           // C1c
  v16h as = frag_const16x32(DM + 512, lane);     // C1s
  v16h bx = load_fragB_s(Xb, 16384, tn * 16, lane);
  v8f ar = {}, ai = {};
  ar = WMMA_F16(ac, bx, ar);
  ai = WMMA_F16(as, bx, ai);
  store_tile(Yr + (size_t)b * 16 * 16384, 16384, 0, tn * 16, lane, ar);
  store_tile(Yi + (size_t)b * 16 * 16384, 16384, 0, tn * 16, lane, ai);
}

// Forward DFT along w: Zr = C2c Yr + C2s Yi ; Zi = C2c Yi - C2s Yr.
__global__ void k_dft_cols(const float* __restrict__ Yr, const float* __restrict__ Yi,
                           const float* __restrict__ DM,
                           float* __restrict__ Zr, float* __restrict__ Zi) {
  int wave = blockIdx.x * 8 + (threadIdx.x >> 5);
  int lane = threadIdx.x & 31;
  int bm = wave >> 5, tc = wave & 31;
  if (bm >= 128) return;
  const float* yr = Yr + (size_t)bm * 32 * 512;
  const float* yi = Yi + (size_t)bm * 32 * 512;
  v16h a_c  = frag_const16x32(DM + 1024, lane);  // C2c
  v16h a_s  = frag_const16x32(DM + 1536, lane);  // +sin
  v16h a_sn = frag_const16x32(DM + 2048, lane);  // -sin
  v16h byr = load_fragB_s(yr, 512, tc * 16, lane);
  v16h byi = load_fragB_s(yi, 512, tc * 16, lane);
  v8f zr = {}, zi = {};
  zr = WMMA_F16(a_c, byr, zr);  zr = WMMA_F16(a_s, byi, zr);
  zi = WMMA_F16(a_c, byi, zi);  zi = WMMA_F16(a_sn, byr, zi);
  store_tile(Zr + (size_t)bm * 16 * 512, 512, 0, tc * 16, lane, zr);
  store_tile(Zi + (size_t)bm * 16 * 512, 512, 0, tc * 16, lane, zi);
}

// ---------------------------------------------------------------------------
// AFNO block-diagonal complex 2-layer MLP, fused (layer1 -> LDS f16 -> layer2).
// Packed weights: mi = ((L*2+p)*8+blk), 8 tiles of 512 halves per matrix.
// ---------------------------------------------------------------------------
__global__ void k_afno_mlp(const float* __restrict__ Zr, const float* __restrict__ Zi,
                           const _Float16* __restrict__ Pw1, const float* __restrict__ ab1,
                           const _Float16* __restrict__ Pw2, const float* __restrict__ ab2,
                           float* __restrict__ O2r, float* __restrict__ O2i, int layer) {
  int blk = blockIdx.x >> 7;
  int rt  = blockIdx.x & 127;
  int wv  = threadIdx.x >> 5;
  int lane = threadIdx.x & 31;
  int part = wv >> 2;                 // 0 real, 1 imag
  int ct = wv & 3;
  int m0 = rt << 4, n0 = ct << 4;
  const size_t mi_r = (size_t)(layer * 2) * 8 + blk;
  const _Float16* W1r = Pw1 + mi_r * 4096;
  const _Float16* W1i = Pw1 + (mi_r + 8) * 4096;
  const _Float16* W2r = Pw2 + mi_r * 4096;
  const _Float16* W2i = Pw2 + (mi_r + 8) * 4096;
  const float* b1 = ab1 + ((size_t)(layer * 2 + part) * 8 + blk) * 64;
  const float* b2 = ab2 + ((size_t)(layer * 2 + part) * 8 + blk) * 64;
  const float* Zr_p = Zr + blk * 64;  // lda 512
  const float* Zi_p = Zi + blk * 64;
  const int abase = (lane < 16) ? 0 : 8;
  const size_t lofs = (size_t)lane * 16;
  __shared__ _Float16 o1s[2][16][64];

  // layer 1:  o1r = gelu(Zr W1r - Zi W1i + b1r); o1i = gelu(Zi W1r + Zr W1i + b1i)
  v8f acc = {};
  for (int kt = 0; kt < 2; ++kt) {
    int k0 = kt << 5;
    size_t arow = (size_t)(m0 + (lane & 15)) * 512 + k0 + abase;
    v16h a1 = cvt_fragA((part ? Zi_p : Zr_p) + arow);
    v16h fb1 = *(const v16h*)(W1r + ((size_t)(kt * 4 + ct) * 32) * 16 + lofs);
    acc = WMMA_F16(a1, fb1, acc);
    v16h a2 = cvt_fragA_s((part ? Zr_p : Zi_p) + arow, part ? 1.0f : -1.0f);
    v16h fb2 = *(const v16h*)(W1i + ((size_t)(kt * 4 + ct) * 32) * 16 + lofs);
    acc = WMMA_F16(a2, fb2, acc);
  }
  {
    int n = n0 + (lane & 15);
    float bb = b1[n];
    int mb = (lane < 16) ? 0 : 8;
#pragma unroll
    for (int r = 0; r < 8; ++r)
      o1s[part][mb + r][n] = (_Float16)gelu_f(acc[r] + bb);
  }
  __syncthreads();

  // layer 2:  o2r = o1r W2r - o1i W2i + b2r; o2i = o1i W2r + o1r W2i + b2i
  v8f acc2 = {};
  for (int kt = 0; kt < 2; ++kt) {
    int k0 = kt << 5;
    v16h a1 = load_fragA_h(&o1s[part][0][0], 64, k0, lane, 1.0f);
    v16h fb1 = *(const v16h*)(W2r + ((size_t)(kt * 4 + ct) * 32) * 16 + lofs);
    acc2 = WMMA_F16(a1, fb1, acc2);
    v16h a2 = load_fragA_h(&o1s[part ^ 1][0][0], 64, k0, lane, part ? 1.0f : -1.0f);
    v16h fb2 = *(const v16h*)(W2i + ((size_t)(kt * 4 + ct) * 32) * 16 + lofs);
    acc2 = WMMA_F16(a2, fb2, acc2);
  }
  {
    float* Oo = part ? O2i : O2r;
    int n = n0 + (lane & 15);
    float bb = b2[n];
    int mb = m0 + ((lane < 16) ? 0 : 8);
#pragma unroll
    for (int r = 0; r < 8; ++r)
      Oo[(size_t)(mb + r) * 512 + blk * 64 + n] = acc2[r] + bb;
  }
}

// Inverse DFT along h: Ur = Er Zr - Ei Zi ; Ui = Er Zi + Ei Zr  (K=16 padded).
__global__ void k_idft_rows(const float* __restrict__ O2r, const float* __restrict__ O2i,
                            const float* __restrict__ DM,
                            float* __restrict__ Ur, float* __restrict__ Ui) {
  int wave = blockIdx.x * 8 + (threadIdx.x >> 5);
  int lane = threadIdx.x & 31;
  if (wave >= 8 * 2 * 512) return;
  int b = wave >> 10, rem = wave & 1023;
  int m0 = (rem >> 9) << 4, n0 = (rem & 511) << 4;
  const float* zr = O2r + (size_t)b * 131072;    // (16 x 8192)
  const float* zi = O2i + (size_t)b * 131072;
  v16h aR  = frag_const32x32(DM + 2560, m0, lane);   // Er padded
  v16h aI  = frag_const32x32(DM + 3584, m0, lane);   // Ei padded
  v16h aIn = frag_const32x32(DM + 4608, m0, lane);   // -Ei padded
  v16h br = load_fragB_sk(zr, 8192, n0, lane, 16);
  v16h bi = load_fragB_sk(zi, 8192, n0, lane, 16);
  v8f ur = {}, ui = {};
  ur = WMMA_F16(aR, br, ur);  ur = WMMA_F16(aIn, bi, ur);
  ui = WMMA_F16(aR, bi, ui);  ui = WMMA_F16(aI, br, ui);
  store_tile(Ur + (size_t)b * 262144, 8192, m0, n0, lane, ur);
  store_tile(Ui + (size_t)b * 262144, 8192, m0, n0, lane, ui);
}

// Inverse rDFT along w + AFNO residual: out = Ar Ur + Ai Ui + skip.
__global__ void k_idft_cols(const float* __restrict__ Ur, const float* __restrict__ Ui,
                            const float* __restrict__ DM,
                            const float* __restrict__ skip, float* __restrict__ Out) {
  int wave = blockIdx.x * 8 + (threadIdx.x >> 5);
  int lane = threadIdx.x & 31;
  if (wave >= 256 * 64) return;
  int bh = wave >> 6, rem = wave & 63;
  int m0 = (rem >> 5) << 4, n0 = (rem & 31) << 4;
  const float* ur = Ur + (size_t)bh * 8192;      // (16 x 512)
  const float* ui = Ui + (size_t)bh * 8192;
  v16h aR = frag_const32x32(DM + 5632, m0, lane);    // Ar padded
  v16h aI = frag_const32x32(DM + 6656, m0, lane);    // Ai padded
  v16h br = load_fragB_sk(ur, 512, n0, lane, 16);
  v16h bi = load_fragB_sk(ui, 512, n0, lane, 16);
  v8f acc = {};
  acc = WMMA_F16(aR, br, acc);  acc = WMMA_F16(aI, bi, acc);
  int n = n0 + (lane & 15);
  int mb = m0 + ((lane < 16) ? 0 : 8);
#pragma unroll
  for (int r = 0; r < 8; ++r) {
    size_t row = (size_t)bh * 32 + mb + r;
    Out[row * 512 + n] = acc[r] + skip[row * 512 + n];
  }
}

// ---------------------------------------------------------------------------
// Data-movement / small kernels
// ---------------------------------------------------------------------------
__global__ void k_im2col(const float* __restrict__ X, float* __restrict__ Xp) {
  int idx = blockIdx.x * 256 + threadIdx.x;
  if (idx >= 32768 * 448) return;
  int k = idx % 448, r = idx / 448;
  int gw = r & 31, gh = (r >> 5) & 31, t = (r >> 10) & 3, b = r >> 12;
  int c = k % 7, pw = (k / 7) & 7, ph = k / 56;
  int yy = gh * 8 + ph, xx = gw * 8 + pw;
  float v;
  if (c < 4)       v = X[((((size_t)b * 256 + yy) * 256 + xx) * 4 + t) * 4 + c];
  else if (c == 4) v = (float)yy * (1.0f / 255.0f);
  else if (c == 5) v = (float)xx * (1.0f / 255.0f);
  else             v = (float)t * (1.0f / 3.0f);
  Xp[idx] = v;
}

__global__ void k_tagg_prep(const float* __restrict__ Hpe, const float* __restrict__ gamma,
                            float* __restrict__ A2) {
  int idx = blockIdx.x * 256 + threadIdx.x;
  if (idx >= 8192 * 2048) return;
  int col = idx & 2047, r = idx >> 11;
  int t = col >> 9, e = col & 511;
  int b = r >> 10, p = r & 1023;
  float te = cosf(((float)t * (1.0f / 3.0f)) * gamma[e]);
  A2[idx] = Hpe[((size_t)(b * 4 + t) * 1024 + p) * 512 + e] * te;
}

__global__ void k_rowmean(const float* __restrict__ H, float* __restrict__ Ct) {
  int b = blockIdx.x, c = threadIdx.x;
  float s = 0.f;
  for (int p = 0; p < 1024; ++p) s += H[((size_t)b * 1024 + p) * 512 + c];
  Ct[b * 512 + c] = s * (1.0f / 1024.0f);
}

__global__ void k_scatter_dec(const float* __restrict__ P, const float* __restrict__ dec_b,
                              float* __restrict__ Y1) {
  int idx = blockIdx.x * 256 + threadIdx.x;
  if (idx >= 8 * 256 * 256 * 32) return;
  int o = idx & 31, x = (idx >> 5) & 255, y = (idx >> 13) & 255, b = idx >> 21;
  int r = b * 1024 + (y >> 3) * 32 + (x >> 3);
  int q = (((y & 7) * 8) + (x & 7)) * 32 + o;
  Y1[idx] = gelu_f(P[(size_t)r * 2048 + q] + dec_b[o]);
}

// 3x3 SAME conv (32->32) + gelu + 1x1 (32->4), fused; weights staged in LDS.
__global__ void k_dec_tail(const float* __restrict__ Y1, const float* __restrict__ c3w,
                           const float* __restrict__ c3b, const float* __restrict__ c1w,
                           const float* __restrict__ c1b, float* __restrict__ Out) {
  __shared__ float s_w3[9 * 32 * 32];
  __shared__ float s_w1[32 * 4];
  __shared__ float s_b3[32], s_b1[4];
  for (int i = threadIdx.x; i < 9216; i += 256) s_w3[i] = c3w[i];
  for (int i = threadIdx.x; i < 128;  i += 256) s_w1[i] = c1w[i];
  if (threadIdx.x < 32) s_b3[threadIdx.x] = c3b[threadIdx.x];
  if (threadIdx.x < 4)  s_b1[threadIdx.x] = c1b[threadIdx.x];
  __syncthreads();
  int idx = blockIdx.x * 256 + threadIdx.x;
  if (idx >= 8 * 256 * 256) return;
  int b = idx >> 16, y = (idx >> 8) & 255, x = idx & 255;
  float acc[32];
#pragma unroll
  for (int o = 0; o < 32; ++o) acc[o] = s_b3[o];
  for (int ky = -1; ky <= 1; ++ky) {
    int yy = y + ky; if (yy < 0 || yy > 255) continue;
    for (int kx = -1; kx <= 1; ++kx) {
      int xx = x + kx; if (xx < 0 || xx > 255) continue;
      const float* p = Y1 + (((size_t)b * 256 + yy) * 256 + xx) * 32;
      const float* wv = &s_w3[(((ky + 1) * 3) + (kx + 1)) * 1024];
      for (int i = 0; i < 32; ++i) {
        float v = p[i];
#pragma unroll
        for (int o = 0; o < 32; ++o) acc[o] += v * wv[i * 32 + o];
      }
    }
  }
  float fin[4] = {s_b1[0], s_b1[1], s_b1[2], s_b1[3]};
  for (int i = 0; i < 32; ++i) {
    float g = gelu_f(acc[i]);
#pragma unroll
    for (int o = 0; o < 4; ++o) fin[o] += g * s_w1[i * 4 + o];
  }
  float* op = Out + (size_t)idx * 4;
#pragma unroll
  for (int o = 0; o < 4; ++o) op[o] = fin[o];
}

// ---------------------------------------------------------------------------
static inline void launch_gemm(const float* A, const _Float16* Bpk, const float* bias,
                               const float* extra, int ermod, int eld, float* C,
                               int M, int N, int Ktiles, int lda, int ldc,
                               int fgelu, hipStream_t s) {
  int waves = ((M + 15) >> 4) * (ntn4_of(N) >> 2);
  k_gemm_pk<<<(waves + 7) / 8, 256, 0, s>>>(A, Bpk, bias, extra, ermod ? ermod : 1,
                                            eld, C, M, N, Ktiles, lda, ldc, fgelu);
}

static inline void launch_pack(const float* Bm, _Float16* out, int K, int N, int ldb,
                               hipStream_t s) {
  int total = (((K + 31) >> 5) * ntn4_of(N)) * 512;
  k_packB<<<(total + 255) / 256, 256, 0, s>>>(Bm, out, K, N, ldb);
}

extern "C" void kernel_launch(void* const* d_in, const int* in_sizes, int n_in,
                              void* d_out, int out_size, void* d_ws, size_t ws_size,
                              hipStream_t stream) {
  (void)in_sizes; (void)n_in; (void)out_size; (void)ws_size;
  const float* x       = (const float*)d_in[0];
  const float* patch_w = (const float*)d_in[1];
  const float* patch_b = (const float*)d_in[2];
  const float* p1_w    = (const float*)d_in[3];
  const float* p1_b    = (const float*)d_in[4];
  const float* pos     = (const float*)d_in[5];
  const float* tagg_w  = (const float*)d_in[6];
  const float* tgamma  = (const float*)d_in[7];
  const float* gn1_s   = (const float*)d_in[8];
  const float* gn1_b   = (const float*)d_in[9];
  const float* aw1     = (const float*)d_in[10];
  const float* ab1     = (const float*)d_in[11];
  const float* aw2     = (const float*)d_in[12];
  const float* ab2     = (const float*)d_in[13];
  const float* gn2_s   = (const float*)d_in[14];
  const float* gn2_b   = (const float*)d_in[15];
  const float* mw1     = (const float*)d_in[16];
  const float* mb1     = (const float*)d_in[17];
  const float* mw2     = (const float*)d_in[18];
  const float* mb2     = (const float*)d_in[19];
  const float* cls_w1  = (const float*)d_in[20];
  const float* cls_b1  = (const float*)d_in[21];
  const float* cls_w2  = (const float*)d_in[22];
  const float* cls_b2  = (const float*)d_in[23];
  const float* cls_w3  = (const float*)d_in[24];
  const float* cls_b3  = (const float*)d_in[25];
  const float* dec_w   = (const float*)d_in[26];
  const float* dec_b   = (const float*)d_in[27];
  const float* c3_w    = (const float*)d_in[28];
  const float* c3_b    = (const float*)d_in[29];
  const float* c1_w    = (const float*)d_in[30];
  const float* c1_b    = (const float*)d_in[31];

  float* wsp = (float*)d_ws;
  size_t off = 0;
  auto alloc = [&](size_t n) { float* r = wsp + off; off += (n + 255) & ~((size_t)255); return r; };
  auto alloch = [&](size_t nh) { return (_Float16*)alloc((nh + 1) / 2); };
  float* DFTM  = alloc(7680);
  float* BIG0  = alloc(16777216);   // im2col -> tagg-A -> decoder patch matrix
  float* HPE   = alloc(16777216);   // patch-embed tokens -> decoder image Y1
  float* H1    = alloc(2097152);    // 32768 x 64 (zero-padded from 35)
  float* HB    = alloc(4194304);
  float* ZB    = alloc(4194304);
  float* Z2    = alloc(4194304);
  float* Z3    = alloc(4194304);
  float* T1    = alloc(4194304);
  float* YR    = alloc(2097152);
  float* YI    = alloc(2097152);
  float* ZRb   = alloc(1048576);
  float* ZIb   = alloc(1048576);
  float* O2R   = alloc(1048576);
  float* O2I   = alloc(1048576);
  float* UR    = alloc(2097152);
  float* UI    = alloc(2097152);
  float* CT0   = alloc(4096);
  float* CT1   = alloc(4096);
  _Float16* P_patch = alloch(14 * 4 * 512);
  _Float16* P_p1    = alloch(2 * 32 * 512);
  _Float16* P_tagg  = alloch(64 * 32 * 512);
  _Float16* P_mw1   = alloch(12 * 16 * 32 * 512);
  _Float16* P_mw2   = alloch(12 * 16 * 32 * 512);
  _Float16* P_cls1  = alloch(16 * 32 * 512);
  _Float16* P_cls2  = alloch(16 * 32 * 512);
  _Float16* P_cls3  = alloch(16 * 4 * 512);
  _Float16* P_dec   = alloch(16 * 128 * 512);
  _Float16* P_aw1   = alloch(192 * 4096);
  _Float16* P_aw2   = alloch(192 * 4096);

  float* yOut  = (float*)d_out;          // (8,256,256,1,4)
  float* cpOut = yOut + 2097152;         // (8,12)

  // constants + weight packing
  k_init_dft<<<1, 1024, 0, stream>>>(DFTM);
  launch_pack(patch_w, P_patch, 448, 35, 35, stream);
  launch_pack(p1_w, P_p1, 35, 512, 512, stream);
  launch_pack(tagg_w, P_tagg, 2048, 512, 512, stream);
  for (int L = 0; L < 12; ++L) {
    launch_pack(mw1 + (size_t)L * 262144, P_mw1 + (size_t)L * 262144, 512, 512, 512, stream);
    launch_pack(mw2 + (size_t)L * 262144, P_mw2 + (size_t)L * 262144, 512, 512, 512, stream);
  }
  launch_pack(cls_w1, P_cls1, 512, 512, 512, stream);
  launch_pack(cls_w2, P_cls2, 512, 512, 512, stream);
  launch_pack(cls_w3, P_cls3, 512, 12, 12, stream);
  k_pack_decw<<<4096, 256, 0, stream>>>(dec_w, P_dec);
  k_pack_afno<<<3072, 256, 0, stream>>>(aw1, P_aw1);
  k_pack_afno<<<3072, 256, 0, stream>>>(aw2, P_aw2);
  k_zero<<<8192, 256, 0, stream>>>(H1, 2097152);

  // patch embed: im2col -> gelu(conv) -> 1x1 + pos
  k_im2col<<<(32768 * 448 + 255) / 256, 256, 0, stream>>>(x, BIG0);
  launch_gemm(BIG0, P_patch, patch_b, nullptr, 0, 0, H1,
              32768, 35, 14, 448, 64, 1, stream);
  launch_gemm(H1, P_p1, p1_b, pos, 1024, 512, HPE,
              32768, 512, 2, 64, 512, 0, stream);

  // exp_mlp time aggregation: (8192 x 2048) @ (2048 x 512)
  k_tagg_prep<<<(8192 * 2048 + 255) / 256, 256, 0, stream>>>(HPE, tgamma, BIG0);
  launch_gemm(BIG0, P_tagg, nullptr, nullptr, 0, 0, HB,
              8192, 512, 64, 2048, 512, 0, stream);

  for (int L = 0; L < 12; ++L) {
    k_groupnorm<<<64, 256, 0, stream>>>(HB, gn1_s + L * 512, gn1_b + L * 512, ZB);
    k_dft_rows<<<1024, 256, 0, stream>>>(ZB, DFTM, YR, YI);
    k_dft_cols<<<512, 256, 0, stream>>>(YR, YI, DFTM, ZRb, ZIb);
    k_afno_mlp<<<1024, 256, 0, stream>>>(ZRb, ZIb, P_aw1, ab1, P_aw2, ab2, O2R, O2I, L);
    k_idft_rows<<<1024, 256, 0, stream>>>(O2R, O2I, DFTM, UR, UI);
    k_idft_cols<<<2048, 256, 0, stream>>>(UR, UI, DFTM, ZB, Z2);
    k_groupnorm<<<64, 256, 0, stream>>>(Z2, gn2_s + L * 512, gn2_b + L * 512, Z3);
    launch_gemm(Z3, P_mw1 + (size_t)L * 262144, mb1 + L * 512, nullptr, 0, 0, T1,
                8192, 512, 16, 512, 512, 1, stream);
    launch_gemm(T1, P_mw2 + (size_t)L * 262144, mb2 + L * 512, HB, 1 << 30, 512, HB,
                8192, 512, 16, 512, 512, 0, stream);
  }

  // classifier head
  k_rowmean<<<8, 512, 0, stream>>>(HB, CT0);
  launch_gemm(CT0, P_cls1, cls_b1, nullptr, 0, 0, CT1, 8, 512, 16, 512, 512, 1, stream);
  launch_gemm(CT1, P_cls2, cls_b2, nullptr, 0, 0, CT0, 8, 512, 16, 512, 512, 1, stream);
  launch_gemm(CT0, P_cls3, cls_b3, nullptr, 0, 0, cpOut, 8, 12, 16, 512, 12, 0, stream);

  // decoder: per-patch projection GEMM -> scatter+gelu -> 3x3 conv + 1x1
  launch_gemm(HB, P_dec, nullptr, nullptr, 0, 0, BIG0,
              8192, 2048, 16, 512, 2048, 0, stream);
  k_scatter_dec<<<(8 * 256 * 256 * 32 + 255) / 256, 256, 0, stream>>>(BIG0, dec_b, HPE);
  k_dec_tail<<<2048, 256, 0, stream>>>(HPE, c3_w, c3_b, c1_w, c1_b, yOut);
}
